// LSTM_15401752723819
// MI455X (gfx1250) — matrix operand order repaired and checked by reading, not compiled, not run on hardware
//
#include <hip/hip_runtime.h>
#include <hip/hip_bf16.h>

// ---------------------------------------------------------------------------
// Problem constants (from reference): B=128, T=1024, IN=256, H=512, G=4H=2048
// ---------------------------------------------------------------------------
#define LB   128
#define LT   1024
#define LIN  256
#define LH   512
#define LG   2048
#define LFC  64

// LDS tile: 16 rows of h_prev, padded to 520 elements (260 DWORDs; 260%64==4
// -> row r starts at bank 4r mod 64, so the 16-lane A-fragment read pattern
// (one 16B chunk per row) covers all 64 banks conflict-free).
#define HS_LD 520

typedef __bf16 bf16_t;
typedef __attribute__((ext_vector_type(16))) __bf16 v16bf;
typedef __attribute__((ext_vector_type(8)))  __bf16 v8bf;
typedef __attribute__((ext_vector_type(8)))  float  v8f;

// ---------------------------------------------------------------------------
// Helpers
// ---------------------------------------------------------------------------
__device__ __forceinline__ v16bf pack16(v8bf lo, v8bf hi) {
    v16bf r;
#pragma unroll
    for (int i = 0; i < 8; ++i) { r[i] = lo[i]; r[i + 8] = hi[i]; }
    return r;
}

__device__ __forceinline__ v8f wmma_bf16(v16bf a, v16bf b, v8f c) {
    // D = A(16x32 bf16) x B(32x16 bf16) + C(16x16 f32)
    return __builtin_amdgcn_wmma_f32_16x16x32_bf16(
        /*neg_a=*/false, a, /*neg_b=*/false, b,
        /*c_mod=*/(short)0, c, /*reuse_a=*/false, /*reuse_b=*/false);
}

__device__ __forceinline__ float fast_sigmoid(float x) {
    return 1.0f / (1.0f + __expf(-x));
}
__device__ __forceinline__ float fast_tanh(float x) {
    float e = __expf(2.0f * x);
    return (e - 1.0f) / (e + 1.0f);
}

// A-fragment (16x32 bf16) per ISA lane map: lanes<16 hold K 0..7 & 16..23,
// lanes>=16 hold K 8..15 & 24..31, row M = lane&15.
__device__ __forceinline__ v16bf load_a_frag(const bf16_t* __restrict__ A,
                                             int mbase, int ldK, int k0,
                                             int lm, int hb) {
    const bf16_t* p = A + (size_t)(mbase + lm) * ldK + k0 + hb * 8;
    v8bf lo = *(const v8bf*)(p);
    v8bf hi = *(const v8bf*)(p + 16);
    return pack16(lo, hi);
}

// B-fragment (32x16 bf16): col N = lane&15; lanes<16 hold K 0..7 & 16..23,
// lanes>=16 hold K 8..15 & 24..31.  B[k][n] = W[n][k] (W row-major N x K).
__device__ __forceinline__ v16bf load_b_frag(const bf16_t* __restrict__ W,
                                             int nrow, int ldK, int k0,
                                             int lm, int hb) {
    const bf16_t* p = W + (size_t)(nrow + lm) * ldK + k0 + hb * 8;
    v8bf lo = *(const v8bf*)(p);
    v8bf hi = *(const v8bf*)(p + 16);
    return pack16(lo, hi);
}

// ---------------------------------------------------------------------------
// Elementwise utility kernels
// ---------------------------------------------------------------------------
__global__ void cvt_f32_bf16(const float* __restrict__ in,
                             bf16_t* __restrict__ out, int n) {
    int i = blockIdx.x * blockDim.x + threadIdx.x;
    if (i < n) out[i] = (bf16_t)in[i];
}

__global__ void bias_sum_kernel(const float* __restrict__ b1,
                                const float* __restrict__ b2,
                                float* __restrict__ out, int n) {
    int i = blockIdx.x * blockDim.x + threadIdx.x;
    if (i < n) out[i] = b1[i] + b2[i];
}

__global__ void zero_state_kernel(bf16_t* __restrict__ h0,
                                  bf16_t* __restrict__ h1,
                                  float* __restrict__ c,
                                  unsigned* __restrict__ bar) {
    int i = blockIdx.x * blockDim.x + threadIdx.x;  // 65536 threads
    if (i < LB * LH) {
        h0[i] = (bf16_t)0.0f;
        h1[i] = (bf16_t)0.0f;
        c[i]  = 0.0f;
    }
    if (i < 2) bar[i] = 0u;
}

// ---------------------------------------------------------------------------
// Large input-projection GEMM:  out[M x 2048] = A[M x K] @ W[2048 x K]^T + bias
// A, W, out in bf16; accumulate f32 in WMMA accumulators.
// One wave computes a 16(M) x 64(N) tile (4 accumulators, A-frag reused 4x).
// grid = (M/16, 4), block = 256 (8 waves, each owning 64 N-columns).
// ---------------------------------------------------------------------------
__global__ __launch_bounds__(256) void gemm_bf16_kernel(
    const bf16_t* __restrict__ A, const bf16_t* __restrict__ W,
    const float* __restrict__ bias, bf16_t* __restrict__ out, int K) {
    const int wid   = threadIdx.x >> 5;
    const int lane  = threadIdx.x & 31;
    const int lm    = lane & 15;
    const int hb    = lane >> 4;
    const int mbase = blockIdx.x * 16;
    const int nb    = (blockIdx.y * 8 + wid) * 64;

    v8f acc[4] = {};

    for (int k0 = 0; k0 < K; k0 += 32) {
        v16bf a = load_a_frag(A, mbase, K, k0, lm, hb);
#pragma unroll
        for (int s = 0; s < 4; ++s) {
            v16bf b = load_b_frag(W, nb + s * 16, K, k0, lm, hb);
            acc[s] = wmma_bf16(a, b, acc[s]);
        }
    }

    // C/D layout: VGPR r, lanes<16 -> M=r, lanes>=16 -> M=r+8; N = lane&15.
#pragma unroll
    for (int r = 0; r < 8; ++r) {
        int m = r + hb * 8;
        size_t row = (size_t)(mbase + m) * LG;
#pragma unroll
        for (int s = 0; s < 4; ++s) {
            int n = nb + s * 16 + lm;
            out[row + n] = (bf16_t)(acc[s][r] + bias[n]);
        }
    }
}

// ---------------------------------------------------------------------------
// Device-wide software barrier (32 co-resident workgroups).
// ---------------------------------------------------------------------------
__device__ __forceinline__ void grid_barrier(unsigned* cnt, unsigned* gen,
                                             unsigned nblk) {
    __threadfence();          // make this block's global writes visible
    __syncthreads();
    if (threadIdx.x == 0) {
        unsigned g = __hip_atomic_load(gen, __ATOMIC_RELAXED,
                                       __HIP_MEMORY_SCOPE_AGENT);
        unsigned a = __hip_atomic_fetch_add(cnt, 1u, __ATOMIC_ACQ_REL,
                                            __HIP_MEMORY_SCOPE_AGENT);
        if (a == nblk - 1u) {
            __hip_atomic_store(cnt, 0u, __ATOMIC_RELAXED,
                               __HIP_MEMORY_SCOPE_AGENT);
            __hip_atomic_store(gen, g + 1u, __ATOMIC_RELEASE,
                               __HIP_MEMORY_SCOPE_AGENT);
        } else {
            while (__hip_atomic_load(gen, __ATOMIC_ACQUIRE,
                                     __HIP_MEMORY_SCOPE_AGENT) == g) {
                __builtin_amdgcn_s_sleep(2);
            }
        }
    }
    __syncthreads();
}

// ---------------------------------------------------------------------------
// Persistent LSTM layer: runs all T=1024 timesteps in one launch.
// Each wave owns a 16(batch) x 16(hidden) output tile and computes the 4 gate
// tiles for it (cols j, j+512, j+1024, j+1536), fused with the cell update.
// Step GEMM: M=128, N=2048, K=512 -> 8 m-tiles x 32 n-tiles = 256 waves =
// 32 blocks x 8 waves.  All 8 waves of a block share the same batch tile, so
// h_prev (16x512, 16 KB) is staged in LDS once per step and A-fragments are
// read with ds_load_b128 (bank-conflict-free via row padding).  w_hh streams
// from L2 (each element used by 8 waves per step, one per batch tile).  h ping-pongs,
// c updates in place, a grid barrier separates timesteps, and the next step's
// xg slice is prefetched behind the pointwise math.
// ---------------------------------------------------------------------------
__global__ __launch_bounds__(256) void lstm_layer_kernel(
    const bf16_t* __restrict__ xg,    // [B*T, 2048] row = b*T + t (bias folded)
    const bf16_t* __restrict__ whh,   // [2048, 512] bf16
    float*  __restrict__ c,           // [B, 512] f32 state
    bf16_t* __restrict__ h0,          // ping
    bf16_t* __restrict__ h1,          // pong
    bf16_t* __restrict__ hseq,        // [B, T, 512] layer output
    unsigned* __restrict__ bar) {
    __shared__ bf16_t hs[16 * HS_LD];  // padded h_prev tile (16.25 KB)

    const int wid   = threadIdx.x >> 5;
    const int lane  = threadIdx.x & 31;
    const int lm    = lane & 15;
    const int hb    = lane >> 4;
    const int mbase = (blockIdx.x >> 2) * 16;              // block-uniform
    const int nbase = (((blockIdx.x & 3) << 3) + wid) * 16;

    for (int t = 0; t < LT; ++t) {
        const bf16_t* hprev = (t & 1) ? h1 : h0;
        bf16_t*       hnext = (t & 1) ? h0 : h1;

        // ---- cooperative LDS stage of the block's h_prev tile (16x512) ----
        // 8192 bf16 = 1024 16B-chunks; 256 threads x 4 chunks.
#pragma unroll
        for (int i = 0; i < 4; ++i) {
            int cid = threadIdx.x + (i << 8);   // 0..1023
            int row = cid >> 6;                 // 64 chunks per row
            int col = (cid & 63) << 3;          // element col, multiple of 8
            *(v8bf*)(&hs[row * HS_LD + col]) =
                *(const v8bf*)(hprev + (size_t)(mbase + row) * LH + col);
        }
        __syncthreads();

        v8f acc[4] = {};  // i, f, g, o gate accumulators (f32)

        for (int k0 = 0; k0 < LH; k0 += 32) {
            // A-fragment from LDS (conflict-free: row r -> banks 4r..4r+3).
            const bf16_t* ap = &hs[lm * HS_LD + k0 + hb * 8];
            v8bf alo = *(const v8bf*)(ap);
            v8bf ahi = *(const v8bf*)(ap + 16);
            v16bf a  = pack16(alo, ahi);
#pragma unroll
            for (int gate = 0; gate < 4; ++gate) {
                v16bf b = load_b_frag(whh, gate * LH + nbase, LH, k0, lm, hb);
                acc[gate] = wmma_bf16(a, b, acc[gate]);
            }
        }

        // Prefetch next step's xg slice for this wave's tile (hides the
        // L2 round-trip behind the pointwise math + barrier).
        if (t + 1 < LT) {
            size_t xrown = ((size_t)(mbase + lm) * LT + (size_t)(t + 1)) * LG;
#pragma unroll
            for (int gate = 0; gate < 4; ++gate) {
                __builtin_prefetch(&xg[xrown + gate * LH + nbase], 0, 3);
            }
        }

        // ---- fused cell update on the 16x16 tile this wave owns ----
#pragma unroll
        for (int r = 0; r < 8; ++r) {
            int m = r + hb * 8;
            int b = mbase + m;
            int j = nbase + lm;
            size_t xrow = ((size_t)b * LT + (size_t)t) * LG;

            float gi = acc[0][r] + (float)xg[xrow + 0 * LH + j];
            float gf = acc[1][r] + (float)xg[xrow + 1 * LH + j];
            float gg = acc[2][r] + (float)xg[xrow + 2 * LH + j];
            float go = acc[3][r] + (float)xg[xrow + 3 * LH + j];

            float iv = fast_sigmoid(gi);
            float fv = fast_sigmoid(gf);
            float gv = fast_tanh(gg);
            float ov = fast_sigmoid(go);

            int cidx = b * LH + j;
            float cv = fv * c[cidx] + iv * gv;
            c[cidx] = cv;
            float hv = ov * fast_tanh(cv);

            hnext[cidx] = (bf16_t)hv;
            hseq[((size_t)b * LT + (size_t)t) * LH + j] = (bf16_t)hv;
        }

        grid_barrier(&bar[0], &bar[1], gridDim.x);
    }
}

// ---------------------------------------------------------------------------
// FC head:  out[b] = sigmoid( relu(h_last @ fc_w^T + fc_b) @ fc2_w^T + fc2_b )
// Tiny (4M MACs): one block, one thread per batch element.
// ---------------------------------------------------------------------------
__global__ void head_kernel(const bf16_t* __restrict__ hseq,
                            const float* __restrict__ fc_w,
                            const float* __restrict__ fc_b,
                            const float* __restrict__ fc2_w,
                            const float* __restrict__ fc2_b,
                            float* __restrict__ out) {
    int b = threadIdx.x;
    if (b >= LB) return;
    const bf16_t* hrow = hseq + ((size_t)b * LT + (LT - 1)) * LH;
    float z2 = fc2_b[0];
    for (int f = 0; f < LFC; ++f) {
        float s = fc_b[f];
        const float* wrow = fc_w + (size_t)f * LH;
        for (int k = 0; k < LH; ++k) s += (float)hrow[k] * wrow[k];
        s = s > 0.0f ? s : 0.0f;
        z2 += s * fc2_w[f];
    }
    out[b] = fast_sigmoid(z2);
}

// ---------------------------------------------------------------------------
// Host launcher
// ---------------------------------------------------------------------------
extern "C" void kernel_launch(void* const* d_in, const int* in_sizes, int n_in,
                              void* d_out, int out_size, void* d_ws,
                              size_t ws_size, hipStream_t stream) {
    const float* x       = (const float*)d_in[0];   // [B,T,IN]
    // d_in[1] length_list: unused by the reference output (h[:, -1, :]).
    const float* w_ih_l0 = (const float*)d_in[2];   // [2048,256]
    const float* w_hh_l0 = (const float*)d_in[3];   // [2048,512]
    const float* b_ih_l0 = (const float*)d_in[4];
    const float* b_hh_l0 = (const float*)d_in[5];
    const float* w_ih_l1 = (const float*)d_in[6];   // [2048,512]
    const float* w_hh_l1 = (const float*)d_in[7];   // [2048,512]
    const float* b_ih_l1 = (const float*)d_in[8];
    const float* b_hh_l1 = (const float*)d_in[9];
    const float* fc_w    = (const float*)d_in[10];
    const float* fc_b    = (const float*)d_in[11];
    const float* fc2_w   = (const float*)d_in[12];
    const float* fc2_b   = (const float*)d_in[13];
    float* out = (float*)d_out;

    // ---- workspace carve (256B aligned slabs) ----
    char* base = (char*)d_ws;
    size_t off = 0;
    auto carve = [&](size_t bytes) -> void* {
        void* p = base + off;
        off += (bytes + 255) & ~(size_t)255;
        return p;
    };
    unsigned* bar  = (unsigned*)carve(256);
    bf16_t* xbf    = (bf16_t*)carve((size_t)LB * LT * LIN * 2);   //  64 MB
    bf16_t* wih0   = (bf16_t*)carve((size_t)LG * LIN * 2);        //   1 MB
    bf16_t* whh0   = (bf16_t*)carve((size_t)LG * LH * 2);         //   2 MB
    bf16_t* wih1   = (bf16_t*)carve((size_t)LG * LH * 2);         //   2 MB
    bf16_t* whh1   = (bf16_t*)carve((size_t)LG * LH * 2);         //   2 MB
    float*  bias0  = (float*)carve((size_t)LG * 4);
    float*  bias1  = (float*)carve((size_t)LG * 4);
    bf16_t* xg     = (bf16_t*)carve((size_t)LB * LT * LG * 2);    // 512 MB
    bf16_t* hseq   = (bf16_t*)carve((size_t)LB * LT * LH * 2);    // 128 MB
    bf16_t* h0     = (bf16_t*)carve((size_t)LB * LH * 2);
    bf16_t* h1     = (bf16_t*)carve((size_t)LB * LH * 2);
    float*  cstate = (float*)carve((size_t)LB * LH * 4);

    const int M = LB * LT;  // 131072 rows for the input-projection GEMMs

    // ---- 1) precision conversion + bias folding ----
    {
        int n = LB * LT * LIN;
        cvt_f32_bf16<<<(n + 255) / 256, 256, 0, stream>>>(x, xbf, n);
        n = LG * LIN;
        cvt_f32_bf16<<<(n + 255) / 256, 256, 0, stream>>>(w_ih_l0, wih0, n);
        n = LG * LH;
        cvt_f32_bf16<<<(n + 255) / 256, 256, 0, stream>>>(w_hh_l0, whh0, n);
        cvt_f32_bf16<<<(n + 255) / 256, 256, 0, stream>>>(w_ih_l1, wih1, n);
        cvt_f32_bf16<<<(n + 255) / 256, 256, 0, stream>>>(w_hh_l1, whh1, n);
        bias_sum_kernel<<<(LG + 255) / 256, 256, 0, stream>>>(b_ih_l0, b_hh_l0,
                                                              bias0, LG);
        bias_sum_kernel<<<(LG + 255) / 256, 256, 0, stream>>>(b_ih_l1, b_hh_l1,
                                                              bias1, LG);
    }

    // ---- 2) layer 0: xg = x @ w_ih0^T + bias0, then persistent recurrence ----
    gemm_bf16_kernel<<<dim3(M / 16, 4), 256, 0, stream>>>(xbf, wih0, bias0, xg,
                                                          LIN);
    zero_state_kernel<<<256, 256, 0, stream>>>(h0, h1, cstate, bar);
    lstm_layer_kernel<<<32, 256, 0, stream>>>(xg, whh0, cstate, h0, h1, hseq,
                                              bar);

    // ---- 3) layer 1: xg = hseq @ w_ih1^T + bias1, then recurrence ----
    gemm_bf16_kernel<<<dim3(M / 16, 4), 256, 0, stream>>>(hseq, wih1, bias1, xg,
                                                          LH);
    zero_state_kernel<<<256, 256, 0, stream>>>(h0, h1, cstate, bar);
    lstm_layer_kernel<<<32, 256, 0, stream>>>(xg, whh1, cstate, h0, h1, hseq,
                                              bar);

    // ---- 4) FC head on h[:, T-1, :] ----
    head_kernel<<<1, 128, 0, stream>>>(hseq, fc_w, fc_b, fc2_w, fc2_b, out);
}
